// MultiheadAttentionLoRA_14998025797851
// MI455X (gfx1250) — matrix operand order repaired and checked by reading, not compiled, not run on hardware
//
#include <hip/hip_runtime.h>

// MI455X (gfx1250) multihead attention:
//   weight cvt -> q/k/v proj (WMMA bf16 GEMM, TDM-staged B, double-buffered)
//   -> flash attention (WMMA) -> out proj (WMMA, TDM-staged A and B)
// B=4, S=2048, D=1024, H=16, HD=64.  Matrix math via v_wmma_f32_16x16x32_bf16.

typedef __bf16 bf16;
typedef __attribute__((ext_vector_type(8)))  bf16  v8bf;
typedef __attribute__((ext_vector_type(16))) bf16  v16bf;
typedef __attribute__((ext_vector_type(8)))  float v8f;
typedef __attribute__((ext_vector_type(4)))  unsigned int u32x4;
typedef __attribute__((ext_vector_type(8)))  int i32x8;
typedef __attribute__((ext_vector_type(4)))  int i32x4;

union Frag {
  v16bf v;
  v8bf  h[2];
};

template <typename T> struct is_bf16      { static constexpr bool value = false; };
template <>           struct is_bf16<bf16>{ static constexpr bool value = true;  };

// group-of-32 ds_swizzle xor (stays inside 16-lane half for mask < 16)
#define SWZ(x, m) \
  __int_as_float(__builtin_amdgcn_ds_swizzle(__float_as_int(x), (((m) << 10) | 0x1f)))

static __device__ __forceinline__ v8f wmma_bf16(const Frag& a, const Frag& b, v8f c) {
  return __builtin_amdgcn_wmma_f32_16x16x32_bf16(
      /*neg_a=*/false, a.v, /*neg_b=*/false, b.v,
      /*c_mod=*/(short)0, c, /*reuse_a=*/false, /*reuse_b=*/false);
}

#if defined(__HIP_DEVICE_COMPILE__) && __has_builtin(__builtin_amdgcn_tensor_load_to_lds)
#define HAVE_TDM 1
#endif

#ifdef HAVE_TDM
// LDS byte offset of a __shared__ object (generic -> AS3 -> int)
#define LDS_OFF(p) \
  ((unsigned)(unsigned long long)(__attribute__((address_space(3))) const void*)(p))

// TDM 2D tile load: tile_w x tile_h bf16 elements, global row stride
// row_stride (elems), destination LDS rows padded by 16B after every 128B.
static __device__ __forceinline__ void tdm_load_tile_2d(
    const bf16* gptr, unsigned lds_byte_off, unsigned tile_w, unsigned tile_h,
    unsigned row_stride) {
  unsigned long long ga = (unsigned long long)(const void*)gptr;
  u32x4 g0;
  g0[0] = 1u;                                        // count=1, user descriptor
  g0[1] = lds_byte_off;                              // lds_addr
  g0[2] = (unsigned)(ga & 0xffffffffu);              // global_addr[31:0]
  g0[3] = (unsigned)((ga >> 32) & 0x01ffffffu)       // global_addr[56:32]
        | (2u << 30);                                // type=2 ("image")
  i32x8 g1;
  g1[0] = (1 << 16)      // data_size = 1 (2 bytes); workgroup_mask = 0
        | (1 << 20)      // pad_enable
        | (4 << 22)      // pad_interval: every 32 DWORDs (128B)
        | (3 << 25);     // pad_amount: 4 DWORDs (16B) -> 72-bf16 LDS rows
  g1[1] = (int)((tile_w & 0xffffu) << 16);           // tensor_dim0[15:0]
  g1[2] = (int)((tile_w >> 16) & 0xffffu)            // tensor_dim0[31:16]
        | (int)((tile_h & 0xffffu) << 16);           // tensor_dim1[15:0]
  g1[3] = (int)((tile_h >> 16) & 0xffffu)            // tensor_dim1[31:16]
        | (int)(tile_w << 16);                       // tile_dim0
  g1[4] = (int)(tile_h & 0xffffu);                   // tile_dim1 (tile_dim2 = 0)
  g1[5] = (int)row_stride;                           // tensor_dim0_stride[31:0]
  g1[6] = 0;                                         // stride hi / dim1_stride lo
  g1[7] = 0;
  const i32x4 z4 = {0, 0, 0, 0};
  const i32x8 z8 = {0, 0, 0, 0, 0, 0, 0, 0};
  __builtin_amdgcn_tensor_load_to_lds(g0, g1, z4, z4, z8, 0);
}
#endif

// ---------------------------------------------------------------------------
// f32 -> bf16 convert (weights)
// ---------------------------------------------------------------------------
__global__ __launch_bounds__(256) void cvt_f32_bf16_kernel(
    const float* __restrict__ src, bf16* __restrict__ dst, int n) {
  const int i = (blockIdx.x * 256 + threadIdx.x) * 4;
  if (i < n) {
    const float4 f = *(const float4*)(src + i);
    dst[i + 0] = (bf16)f.x;
    dst[i + 1] = (bf16)f.y;
    dst[i + 2] = (bf16)f.z;
    dst[i + 3] = (bf16)f.w;
  }
}

// ---------------------------------------------------------------------------
// GEMM: C[M,N] = A[M,K] @ W[N,K]^T + bias, A is AT (f32 or bf16), W is bf16.
// Block tile 128x128, 8 waves, wave tile 32x64, BK=64 (2 wmma k-steps/stage).
// Double-buffered LDS: TDM DMA for slab k+1 is issued before computing slab k,
// and only completed (s_wait_tensorcnt) after the WMMAs for slab k retire.
// ---------------------------------------------------------------------------
constexpr int BM = 128, BN = 128, BK = 64;
constexpr int LDT = BK + 8;  // 72 bf16 = 144B padded row (matches TDM pad)

template <typename AT, typename OT>
__global__ __launch_bounds__(256) void gemm_bias_kernel(
    const AT* __restrict__ X, const bf16* __restrict__ W,
    const float* __restrict__ bias, OT* __restrict__ C,
    int M, int N, int K) {
  __shared__ alignas(16) bf16 As[2][BM][LDT];
  __shared__ alignas(16) bf16 Bs[2][BN][LDT];

  const int tid  = threadIdx.x;
  const int lane = tid & 31;
  const int wid  = tid >> 5;
  const int ml   = lane & 15;
  const int hi   = lane >> 4;
  const int wm   = (wid & 3) * 32;
  const int wn   = (wid >> 2) * 64;
  const int m0   = blockIdx.y * BM;
  const int n0   = blockIdx.x * BN;

  const int srow = tid >> 1;        // 0..127
  const int scol = (tid & 1) * 32;  // 0 / 32

  // stage one k-slab into LDS buffer `buf` (TDM for bf16 operands,
  // fused f32->bf16 convert for the f32 A operand)
  auto stage = [&](int buf, int k0) {
#ifdef HAVE_TDM
    if (wid == 0) {  // wave-level DMA issue (EXEC ignored by TENSOR ops)
      tdm_load_tile_2d(W + (size_t)n0 * K + k0, LDS_OFF(&Bs[buf][0][0]), BK, BN, K);
      if constexpr (is_bf16<AT>::value)
        tdm_load_tile_2d((const bf16*)X + (size_t)m0 * K + k0,
                         LDS_OFF(&As[buf][0][0]), BK, BM, K);
    }
    if constexpr (!is_bf16<AT>::value) {
      const AT* ap = X + (size_t)(m0 + srow) * K + k0 + scol;
#pragma unroll
      for (int i = 0; i < 32; ++i) As[buf][srow][scol + i] = (bf16)(float)ap[i];
      if (k0 + BK < K) __builtin_prefetch(ap + BK, 0, 0);
    }
#else
    const AT*   ap = X + (size_t)(m0 + srow) * K + k0 + scol;
    const bf16* wp = W + (size_t)(n0 + srow) * K + k0 + scol;
#pragma unroll
    for (int i = 0; i < 32; ++i) As[buf][srow][scol + i] = (bf16)(float)ap[i];
#pragma unroll
    for (int i = 0; i < 32; ++i) Bs[buf][srow][scol + i] = wp[i];
    if (k0 + BK < K) {
      __builtin_prefetch(ap + BK, 0, 0);
      __builtin_prefetch(wp + BK, 0, 0);
    }
#endif
  };

  v8f acc[2][4];
#pragma unroll
  for (int i = 0; i < 2; ++i)
#pragma unroll
    for (int j = 0; j < 4; ++j) acc[i][j] = {0.f, 0.f, 0.f, 0.f, 0.f, 0.f, 0.f, 0.f};

  // prologue: fill buffer 0
  stage(0, 0);
#ifdef HAVE_TDM
  if (wid == 0) __builtin_amdgcn_s_wait_tensorcnt(0);
#endif
  __syncthreads();

  const int NK = K / BK;
  for (int kt = 0; kt < NK; ++kt) {
    const int cur = kt & 1;
    if (kt + 1 < NK) stage(cur ^ 1, (kt + 1) * BK);  // overlap DMA with WMMAs

#pragma unroll
    for (int ks = 0; ks < BK; ks += 32) {
      Frag a[2], b[4];
#pragma unroll
      for (int i = 0; i < 2; ++i) {  // A 16x32: v0-3 K=hi*8.., v4-7 K=16+hi*8..
        const bf16* p = &As[cur][wm + i * 16 + ml][ks + hi * 8];
        a[i].h[0] = *(const v8bf*)p;
        a[i].h[1] = *(const v8bf*)(p + 16);
      }
#pragma unroll
      for (int j = 0; j < 4; ++j) {  // B 32x16: lane=N, v0-3 K=hi*8.., v4-7 K=16+hi*8..
        const bf16* p = &Bs[cur][wn + j * 16 + ml][ks + hi * 8];
        b[j].h[0] = *(const v8bf*)p;
        b[j].h[1] = *(const v8bf*)(p + 16);
      }
#pragma unroll
      for (int i = 0; i < 2; ++i)
#pragma unroll
        for (int j = 0; j < 4; ++j) acc[i][j] = wmma_bf16(a[i], b[j], acc[i][j]);
    }

#ifdef HAVE_TDM
    if (kt + 1 < NK && wid == 0) __builtin_amdgcn_s_wait_tensorcnt(0);
#endif
    __syncthreads();
  }

  // epilogue: C/D layout -> row = hi*8 + r, col = ml
#pragma unroll
  for (int i = 0; i < 2; ++i) {
#pragma unroll
    for (int j = 0; j < 4; ++j) {
      const int   col = n0 + wn + j * 16 + ml;
      const float bv  = bias[col];
#pragma unroll
      for (int r = 0; r < 8; ++r) {
        const int row = m0 + wm + i * 16 + hi * 8 + r;
        C[(size_t)row * N + col] = (OT)(acc[i][j][r] + bv);
      }
    }
  }
}

// ---------------------------------------------------------------------------
// Flash attention.  grid = (B*H, S/64), block = 128 (4 waves x 16 q-rows).
// Q/K/V/ctx are bf16 [B*S, D]; head h lives in columns h*64..h*64+63.
// ---------------------------------------------------------------------------
constexpr int ATT_S = 2048, ATT_D = 1024, HD = 64;
constexpr int VLD = 72;

__global__ __launch_bounds__(128) void flash_attn_kernel(
    const bf16* __restrict__ Q, const bf16* __restrict__ Km,
    const bf16* __restrict__ V, bf16* __restrict__ ctx) {
  __shared__ alignas(16) bf16 VT[HD][VLD];     // V^T chunk: [d][s]
  __shared__ alignas(16) bf16 Pl[4][16][VLD];  // per-wave P tile (D->A layout hop)

  const int tid  = threadIdx.x;
  const int lane = tid & 31;
  const int wid  = tid >> 5;
  const int ml   = lane & 15;
  const int hi   = lane >> 4;

  const int b  = blockIdx.x >> 4;
  const int h  = blockIdx.x & 15;
  const int q0 = blockIdx.y * 64 + wid * 16;

  const size_t rowbase = (size_t)b * ATT_S;
  const size_t hoff    = (size_t)h * HD;

  // Q fragments (16 rows x 64 head-dim) stay in registers for the whole S loop
  Frag qa[2];
  {
    const bf16* qp = Q + (rowbase + q0 + ml) * ATT_D + hoff;
#pragma unroll
    for (int i = 0; i < 2; ++i) {
      qa[i].h[0] = *(const v8bf*)(qp + i * 32 + hi * 8);
      qa[i].h[1] = *(const v8bf*)(qp + i * 32 + 16 + hi * 8);
    }
  }

  float mrow[8], lrow[8];
  v8f   o[4];
#pragma unroll
  for (int r = 0; r < 8; ++r) { mrow[r] = -3.0e38f; lrow[r] = 0.f; }
#pragma unroll
  for (int j = 0; j < 4; ++j) o[j] = {0.f, 0.f, 0.f, 0.f, 0.f, 0.f, 0.f, 0.f};

  for (int s0 = 0; s0 < ATT_S; s0 += 64) {
    __syncthreads();
    // stage V chunk transposed: VT[d][s] = V[s0+s][h*64+d]
#pragma unroll
    for (int it = 0; it < 4; ++it) {
      const int idx = tid + it * 128;  // 512 chunks of 8 bf16
      const int sr  = idx >> 3;
      const int d0  = (idx & 7) * 8;
      v8bf t = *(const v8bf*)(V + (rowbase + s0 + sr) * ATT_D + hoff + d0);
#pragma unroll
      for (int j = 0; j < 8; ++j) VT[d0 + j][sr] = t[j];
    }
    __syncthreads();

    // scores S = Q K^T  (16 x 64), K B-fragments straight from global
    v8f sc[4];
#pragma unroll
    for (int nt = 0; nt < 4; ++nt) {
      const bf16* kp = Km + (rowbase + s0 + nt * 16 + ml) * ATT_D + hoff;
      Frag kb0, kb1;
      kb0.h[0] = *(const v8bf*)(kp + hi * 8);
      kb0.h[1] = *(const v8bf*)(kp + 16 + hi * 8);
      kb1.h[0] = *(const v8bf*)(kp + 32 + hi * 8);
      kb1.h[1] = *(const v8bf*)(kp + 32 + 16 + hi * 8);
      v8f s = {0.f, 0.f, 0.f, 0.f, 0.f, 0.f, 0.f, 0.f};
      s = wmma_bf16(qa[0], kb0, s);
      s = wmma_bf16(qa[1], kb1, s);
      sc[nt] = s;
    }

    // online softmax: row (this half-wave) = hi*8 + r, reduce across 16 lanes
    const float scale = 0.125f;  // 1/sqrt(64)
    float mnew[8], corr[8], rsum[8];
#pragma unroll
    for (int r = 0; r < 8; ++r) {
      float v0 = fmaxf(fmaxf(sc[0][r], sc[1][r]), fmaxf(sc[2][r], sc[3][r]));
      v0 = fmaxf(v0, SWZ(v0, 1));
      v0 = fmaxf(v0, SWZ(v0, 2));
      v0 = fmaxf(v0, SWZ(v0, 4));
      v0 = fmaxf(v0, SWZ(v0, 8));
      mnew[r] = fmaxf(mrow[r], v0 * scale);
      corr[r] = __expf(mrow[r] - mnew[r]);
      rsum[r] = 0.f;
    }
#pragma unroll
    for (int nt = 0; nt < 4; ++nt) {
#pragma unroll
      for (int r = 0; r < 8; ++r) {
        float p = __expf(sc[nt][r] * scale - mnew[r]);
        rsum[r] += p;
        Pl[wid][hi * 8 + r][nt * 16 + ml] = (bf16)p;  // D-layout -> row-major LDS
      }
    }
#pragma unroll
    for (int r = 0; r < 8; ++r) {
      float t = rsum[r];
      t += SWZ(t, 1);
      t += SWZ(t, 2);
      t += SWZ(t, 4);
      t += SWZ(t, 8);
      lrow[r] = lrow[r] * corr[r] + t;
      mrow[r] = mnew[r];
    }
#pragma unroll
    for (int j = 0; j < 4; ++j)
#pragma unroll
      for (int r = 0; r < 8; ++r) o[j][r] *= corr[r];

    // cross-lane LDS dependency within the wave: drain DS before re-reading
    asm volatile("s_wait_dscnt 0" ::: "memory");

    // P as A-matrix (16 x 64), V^T columns as B-matrix
    Frag pa[2];
    {
      const bf16* pp = &Pl[wid][ml][0];
#pragma unroll
      for (int i = 0; i < 2; ++i) {
        pa[i].h[0] = *(const v8bf*)(pp + i * 32 + hi * 8);
        pa[i].h[1] = *(const v8bf*)(pp + i * 32 + 16 + hi * 8);
      }
    }
#pragma unroll
    for (int dt = 0; dt < 4; ++dt) {
      const bf16* vp = &VT[dt * 16 + ml][0];
      Frag vb0, vb1;
      vb0.h[0] = *(const v8bf*)(vp + hi * 8);
      vb0.h[1] = *(const v8bf*)(vp + 16 + hi * 8);
      vb1.h[0] = *(const v8bf*)(vp + 32 + hi * 8);
      vb1.h[1] = *(const v8bf*)(vp + 32 + 16 + hi * 8);
      o[dt] = wmma_bf16(pa[0], vb0, o[dt]);
      o[dt] = wmma_bf16(pa[1], vb1, o[dt]);
    }
  }

  float inv[8];
#pragma unroll
  for (int r = 0; r < 8; ++r) inv[r] = 1.f / lrow[r];
#pragma unroll
  for (int dt = 0; dt < 4; ++dt)
#pragma unroll
    for (int r = 0; r < 8; ++r) {
      ctx[(rowbase + q0 + hi * 8 + r) * ATT_D + hoff + dt * 16 + ml] =
          (bf16)(o[dt][r] * inv[r]);
    }
}

// ---------------------------------------------------------------------------
extern "C" void kernel_launch(void* const* d_in, const int* in_sizes, int n_in,
                              void* d_out, int out_size, void* d_ws, size_t ws_size,
                              hipStream_t stream) {
  const float* query = (const float*)d_in[0];
  const float* key   = (const float*)d_in[1];
  const float* value = (const float*)d_in[2];
  const float* Wq = (const float*)d_in[3];
  const float* bq = (const float*)d_in[4];
  const float* Wk = (const float*)d_in[5];
  const float* bk = (const float*)d_in[6];
  const float* Wv = (const float*)d_in[7];
  const float* bv = (const float*)d_in[8];
  const float* Wo = (const float*)d_in[9];
  const float* bo = (const float*)d_in[10];
  float* out = (float*)d_out;

  const int Mrows = 4 * 2048;  // B * S
  const int D = 1024;
  const size_t NE = (size_t)Mrows * D;
  const size_t WE = (size_t)D * D;

  // workspace: 4x bf16 [8192,1024] + 4x bf16 [1024,1024] (~75 MB)
  bf16* Qp  = (bf16*)d_ws;
  bf16* Kp  = Qp + NE;
  bf16* Vp  = Kp + NE;
  bf16* Ctx = Vp + NE;
  bf16* Wqb = Ctx + NE;
  bf16* Wkb = Wqb + WE;
  bf16* Wvb = Wkb + WE;
  bf16* Wob = Wvb + WE;

  // weight conversion (f32 -> bf16)
  {
    dim3 cg((int)(WE / (256 * 4))), cb(256);
    cvt_f32_bf16_kernel<<<cg, cb, 0, stream>>>(Wq, Wqb, (int)WE);
    cvt_f32_bf16_kernel<<<cg, cb, 0, stream>>>(Wk, Wkb, (int)WE);
    cvt_f32_bf16_kernel<<<cg, cb, 0, stream>>>(Wv, Wvb, (int)WE);
    cvt_f32_bf16_kernel<<<cg, cb, 0, stream>>>(Wo, Wob, (int)WE);
  }

  dim3 g(D / BN, Mrows / BM), blk(256);
  gemm_bias_kernel<float, bf16><<<g, blk, 0, stream>>>(query, Wqb, bq, Qp, Mrows, D, D);
  gemm_bias_kernel<float, bf16><<<g, blk, 0, stream>>>(key,   Wkb, bk, Kp, Mrows, D, D);
  gemm_bias_kernel<float, bf16><<<g, blk, 0, stream>>>(value, Wvb, bv, Vp, Mrows, D, D);

  flash_attn_kernel<<<dim3(4 * 16, ATT_S / 64), 128, 0, stream>>>(Qp, Kp, Vp, Ctx);

  gemm_bias_kernel<bf16, float><<<g, blk, 0, stream>>>(Ctx, Wob, bo, out, Mrows, D, D);
}
